// SelfAttention_13134009991753
// MI455X (gfx1250) — matrix operand order repaired and checked by reading, not compiled, not run on hardware
//
#include <hip/hip_runtime.h>
#include <stdint.h>

typedef _Float16 half_t;
typedef __attribute__((ext_vector_type(16))) _Float16 v16h;
typedef __attribute__((ext_vector_type(8)))  _Float16 v8h;
typedef __attribute__((ext_vector_type(8)))  float    v8f;
typedef __attribute__((ext_vector_type(4)))  int      v4i;

#define BB 4
#define CC 256
#define HH 128
#define WW 128

// LDS row strides (in halves), padded so 16 rows hit 16 distinct 16B bank
// groups (stride*2 bytes ≡ 16 mod 256).
#define LKP 264   // K tile rows (256 + 8)
#define LVP 136   // V^T tile rows (128 + 8)
#define LPP 136   // P tile rows (128 + 8)

#define HAS_ASYNC_LDS __has_builtin(__builtin_amdgcn_global_load_async_to_lds_b128)

typedef __attribute__((address_space(1))) v4i gbl_v4i;
typedef __attribute__((address_space(3))) v4i lds_v4i;

__device__ __forceinline__ gbl_v4i* to_gbl(const void* p) {
  return (gbl_v4i*)(uintptr_t)p;  // AS1 pointers carry the full 64-bit VA
}
__device__ __forceinline__ lds_v4i* to_lds(void* p) {
  // AS3 pointers are 32-bit LDS byte offsets == low bits of the flat address.
  return (lds_v4i*)(uint32_t)(uintptr_t)p;
}

// ---------------------------------------------------------------------------
// Pass 1: strided transpose + f32 -> f16 convert through an LDS tile.
// src slice s: base = (s/div)*srcA + (s%div)*srcB, rows y stride srcRow,
// inner dim x contiguous (coalesced reads along x).
// dst: base = (s/div)*dstA + (s%div)*dstB, row x stride dstRow, inner dim y
// contiguous (coalesced writes along y).
// ---------------------------------------------------------------------------
__global__ __launch_bounds__(256) void transpose_cvt(
    const float* __restrict__ src, half_t* __restrict__ dst,
    int div_, long srcA, long srcB, long srcRow,
    long dstA, long dstB, long dstRow) {
  __shared__ half_t tile[32][34];
  const int s = blockIdx.z;
  const long sbase = (long)(s / div_) * srcA + (long)(s % div_) * srcB;
  const long dbase = (long)(s / div_) * dstA + (long)(s % div_) * dstB;
  const int x0 = blockIdx.x * 32;
  const int y0 = blockIdx.y * 32;
  const int tx = threadIdx.x;   // 0..31
  const int ty = threadIdx.y;   // 0..7
#pragma unroll
  for (int j = 0; j < 4; ++j) {
    int y = y0 + ty + j * 8;
    tile[ty + j * 8][tx] = (half_t)src[sbase + (long)y * srcRow + (x0 + tx)];
  }
  __syncthreads();
#pragma unroll
  for (int j = 0; j < 4; ++j) {
    int x = x0 + ty + j * 8;
    dst[dbase + (long)x * dstRow + (y0 + tx)] = tile[tx][ty + j * 8];
  }
}

__device__ __forceinline__ v16h cat16(v8h lo, v8h hi) {
  return __builtin_shufflevector(lo, hi, 0, 1, 2, 3, 4, 5, 6, 7,
                                 8, 9, 10, 11, 12, 13, 14, 15);
}

// ---------------------------------------------------------------------------
// Pass 2: one workgroup per (b,w) block. 8 waves x 32 lanes.
// Qh/Kh: (B,W,H,C) f16 contiguous per block; Vh: (B,W,C,H) f16.
// Wave m-tile: rows moff..moff+15 of Q. S = Q K^T / 16; P = softmax(S);
// O = P V, written to out in (B,W,H,C) order == raw reshape of reference.
// ---------------------------------------------------------------------------
__global__ __launch_bounds__(256) void attn_kernel(
    const half_t* __restrict__ Qh, const half_t* __restrict__ Kh,
    const half_t* __restrict__ Vh, float* __restrict__ out) {
  // K tile reuses its space for P after GEMM1 (barrier-separated).
  __shared__ half_t smem[HH * LKP + CC * LVP];
  half_t* Klds = smem;
  half_t* Vlds = smem + HH * LKP;
  half_t* Plds = smem;  // alias of Klds, valid after post-GEMM1 barrier

  const int blk  = blockIdx.x;      // b*W + w
  const int tid  = threadIdx.x;
  const int lane = tid & 31;
  const int wave = tid >> 5;        // 0..7
  const int hf   = (lane >> 4) & 1; // lane half (0: lanes 0-15, 1: 16-31)
  const int l15  = lane & 15;

  const long tileOff = (long)blk * (HH * CC);  // 32768 halves per block
  const half_t* Qg = Qh + tileOff;
  const half_t* Kg = Kh + tileOff;
  const half_t* Vg = Vh + tileOff;

  // Stage K (128x256) and V^T (256x128) tiles into LDS.
#if HAS_ASYNC_LDS
  // CDNA5 async copy: memory -> LDS directly, no VGPR round trip; tracked by
  // ASYNCcnt, overlaps with the Q fragment loads below.
#pragma unroll
  for (int i = 0; i < 16; ++i) {
    int idx = (i * 256 + tid) * 8;  // half index, 16B chunks
    __builtin_amdgcn_global_load_async_to_lds_b128(
        to_gbl(&Kg[idx]), to_lds(&Klds[(idx >> 8) * LKP + (idx & 255)]), 0, 0);
    __builtin_amdgcn_global_load_async_to_lds_b128(
        to_gbl(&Vg[idx]), to_lds(&Vlds[(idx >> 7) * LVP + (idx & 127)]), 0, 0);
  }
#else
#pragma unroll
  for (int i = 0; i < 16; ++i) {
    int idx = (i * 256 + tid) * 8;  // half index, 16B-aligned chunks
    *(v8h*)&Klds[(idx >> 8) * LKP + (idx & 255)] = *(const v8h*)&Kg[idx];
    *(v8h*)&Vlds[(idx >> 7) * LVP + (idx & 127)] = *(const v8h*)&Vg[idx];
  }
#endif

  // Q A-fragments (16x32 each) straight from global into registers.
  // A layout: lanes 0-15 m=lane, halves 0-7 = K 0-7, halves 8-15 = K 16-23;
  // lanes 16-31: K 8-15 / K 24-31.
  const int moff = wave * 16;
  v16h qf[8];
  {
    const half_t* qrow = Qg + (moff + l15) * CC;
#pragma unroll
    for (int ks = 0; ks < 8; ++ks) {
      v8h lo = *(const v8h*)&qrow[ks * 32 + hf * 8];
      v8h hi = *(const v8h*)&qrow[ks * 32 + 16 + hf * 8];
      qf[ks] = cat16(lo, hi);
    }
  }

#if HAS_ASYNC_LDS
  // Drain this wave's async LDS writes before the workgroup barrier.
#if __has_builtin(__builtin_amdgcn_s_wait_asynccnt)
  __builtin_amdgcn_s_wait_asynccnt(0);
#else
  asm volatile("s_wait_asynccnt 0" ::: "memory");
#endif
#endif
  __syncthreads();

  // GEMM1: S(16x128) = Q(16x256) * K^T. B[k=c][n=h_k] = K[h_k][c]:
  // lane col n = l15 -> K row nt*16+l15; c at hf*8+0..7 and 16+hf*8+0..7.
  v8f sacc[8];
#pragma unroll
  for (int nt = 0; nt < 8; ++nt) {
    v8f acc = {};
#pragma unroll
    for (int ks = 0; ks < 8; ++ks) {
      const half_t* bp = &Klds[(nt * 16 + l15) * LKP + ks * 32 + hf * 8];
      v16h bf = cat16(*(const v8h*)bp, *(const v8h*)(bp + 16));
      acc = __builtin_amdgcn_wmma_f32_16x16x32_f16(false, qf[ks], false, bf,
                                                   (short)0, acc, false, false);
    }
    sacc[nt] = acc;
  }

  __syncthreads();  // everyone done reading Klds; safe to overwrite with P

  // Softmax. D fragment: VGPR r holds row (moff + r + 8*hf), col l15.
  // Row reduction = 8 frags per lane + xor-shuffles within 16-lane half.
  const float inv_scale = 0.0625f;  // 1/sqrt(256)
#pragma unroll
  for (int r = 0; r < 8; ++r) {
    float mx = -3.0e38f;
#pragma unroll
    for (int nt = 0; nt < 8; ++nt) mx = fmaxf(mx, sacc[nt][r]);
#pragma unroll
    for (int m = 8; m >= 1; m >>= 1) mx = fmaxf(mx, __shfl_xor(mx, m, 32));
    float sum = 0.0f;
#pragma unroll
    for (int nt = 0; nt < 8; ++nt) {
      float p = __expf((sacc[nt][r] - mx) * inv_scale);
      sacc[nt][r] = p;
      sum += p;
    }
#pragma unroll
    for (int m = 8; m >= 1; m >>= 1) sum += __shfl_xor(sum, m, 32);
    const float rn = 1.0f / sum;
    const int prow = moff + r + 8 * hf;
#pragma unroll
    for (int nt = 0; nt < 8; ++nt)
      Plds[prow * LPP + nt * 16 + l15] = (half_t)(sacc[nt][r] * rn);
  }
  // GEMM2's A-fragments only read this wave's own P rows -> no extra barrier.

  // GEMM2: O(16x256) = P(16x128) * V. B[k=h][n=c] = Vt[c][h], h contiguous.
  float* orow = out + tileOff;
#pragma unroll
  for (int nt = 0; nt < 16; ++nt) {
    v8f acc = {};
#pragma unroll
    for (int ks = 0; ks < 4; ++ks) {
      const half_t* ap = &Plds[(moff + l15) * LPP + ks * 32 + hf * 8];
      v16h af = cat16(*(const v8h*)ap, *(const v8h*)(ap + 16));
      const half_t* bp = &Vlds[(nt * 16 + l15) * LVP + ks * 32 + hf * 8];
      v16h bf = cat16(*(const v8h*)bp, *(const v8h*)(bp + 16));
      acc = __builtin_amdgcn_wmma_f32_16x16x32_f16(false, af, false, bf,
                                                   (short)0, acc, false, false);
    }
#pragma unroll
    for (int r = 0; r < 8; ++r)
      orow[(moff + r + 8 * hf) * CC + nt * 16 + l15] = acc[r];
  }
}

// ---------------------------------------------------------------------------
extern "C" void kernel_launch(void* const* d_in, const int* in_sizes, int n_in,
                              void* d_out, int out_size, void* d_ws, size_t ws_size,
                              hipStream_t stream) {
  const float* Q = (const float*)d_in[0];
  const float* K = (const float*)d_in[1];
  const float* V = (const float*)d_in[2];
  float* outp = (float*)d_out;

  const long N = (long)BB * CC * HH * WW;  // 16,777,216 elems
  // Workspace: 3 * N * 2 bytes = 96 MB of f16 transposed tensors.
  half_t* Qh = (half_t*)d_ws;
  half_t* Kh = Qh + N;
  half_t* Vh = Kh + N;

  dim3 tb(32, 8, 1);

  // Q/K: (B,C,H,W) -> (B,W,H,C). Slice s = b*H + h; y-dim = c (rows C),
  // x-dim = w (cols W).
  dim3 gqk(WW / 32, CC / 32, BB * HH);
  transpose_cvt<<<gqk, tb, 0, stream>>>(
      Q, Qh, HH,
      (long)CC * HH * WW, (long)WW, (long)HH * WW,
      (long)WW * HH * CC, (long)CC, (long)HH * CC);
  transpose_cvt<<<gqk, tb, 0, stream>>>(
      K, Kh, HH,
      (long)CC * HH * WW, (long)WW, (long)HH * WW,
      (long)WW * HH * CC, (long)CC, (long)HH * CC);

  // V: (B,C,H,W) -> (B,W,C,H). Slice s = b*C + c; y-dim = h (rows H),
  // x-dim = w (cols W).
  dim3 gv(WW / 32, HH / 32, BB * CC);
  transpose_cvt<<<gv, tb, 0, stream>>>(
      V, Vh, CC,
      (long)CC * HH * WW, (long)HH * WW, (long)WW,
      (long)WW * CC * HH, (long)HH, (long)CC * HH);

  // One workgroup per (b,w) attention block.
  attn_kernel<<<BB * WW, 256, 0, stream>>>(Qh, Kh, Vh, outp);
}
